// MERITS_Lin_63934883169132
// MI455X (gfx1250) — matrix operand order repaired and checked by reading, not compiled, not run on hardware
//
#include <hip/hip_runtime.h>
#include <math.h>
#include <stdint.h>

// ---------------------------------------------------------------------------
// Types for CDNA5 WMMA (wave32): D(f32 16x16) = A(bf16 16x32) x B(bf16 32x16) + C
// ---------------------------------------------------------------------------
typedef __bf16 bf16;
typedef __bf16 v16bf  __attribute__((ext_vector_type(16)));
typedef __bf16 bf16x2 __attribute__((ext_vector_type(2)));
typedef float  v8f    __attribute__((ext_vector_type(8)));
typedef uint32_t u32;
typedef u32 u32x2 __attribute__((ext_vector_type(2)));
typedef u32 u32x4 __attribute__((ext_vector_type(4)));

union Frag {            // 8 dwords = 16 packed bf16 = one WMMA operand
    u32x4 q[2];
    v16bf v;
};

__device__ __forceinline__ u32 pack_bf16(float lo, float hi) {
    // even K in [15:0], odd K in [31:16]; native bf16x2 -> single v_cvt_pk_bf16_f32
    bf16x2 p = { (__bf16)lo, (__bf16)hi };
    return __builtin_bit_cast(u32, p);
}

#define BM 64            // block tile M (4 WMMA tiles)
#define BN 64            // block tile N (one 16-col slab per wave)
#define KT 32            // K slab = WMMA K depth
#define LDP 20           // padded row length in u32 (80B: 16B-aligned, conflict-breaking)

// C[M,N] = act( (RELU_IN? relu(A) : A)[M,K] @ W[K,N] + bias[N] )
// A,W,C row-major dense.  REQUIRES: M%64==0, N%64==0, K%32==0 (all call sites do).
// ACT: 0=none, 1=relu, 2=tanh.
template<int ACT, bool RELU_IN>
__global__ __launch_bounds__(128)
void gemm_wmma_bf16(const float* __restrict__ A, const float* __restrict__ W,
                    const float* __restrict__ bias, float* __restrict__ C,
                    int M, int N, int K)
{
    __shared__ __align__(16) u32 Apk[BM][LDP];   // Apk[row][kpair]
    __shared__ __align__(16) u32 Bpk[BN][LDP];   // Bpk[col][kpair]

    const int tid   = threadIdx.x;
    const int wave  = tid >> 5;
    const int lane  = tid & 31;
    const int m_blk = blockIdx.x * BM;
    const int n_blk = blockIdx.y * BN;

    const int h     = lane >> 4;        // half-wave select
    const int r16   = lane & 15;        // row (A) / col (B,C) within 16
    const int ncol  = wave * 16 + r16;  // block-local output column

    v8f acc[4];
    #pragma unroll
    for (int mt = 0; mt < 4; ++mt)
        acc[mt] = (v8f){0.f,0.f,0.f,0.f,0.f,0.f,0.f,0.f};

    for (int k0 = 0; k0 < K; k0 += KT) {
        // ---- A tile: 64 rows x 8 float4 groups, coalesced b128 loads,
        //      two packed pairs stored as one b64
        #pragma unroll
        for (int u = 0; u < 4; ++u) {
            int p = tid + u * 128;
            int r = p >> 3, kq = p & 7;
            float4 t = *(const float4*)(A + (size_t)(m_blk + r) * K + k0 + 4 * kq);
            if (RELU_IN) {
                t.x = fmaxf(t.x, 0.f); t.y = fmaxf(t.y, 0.f);
                t.z = fmaxf(t.z, 0.f); t.w = fmaxf(t.w, 0.f);
            }
            u32x2 pr = { pack_bf16(t.x, t.y), pack_bf16(t.z, t.w) };
            *(u32x2*)&Apk[r][2 * kq] = pr;
        }
        // ---- B tile: 16 kpairs x 16 col-groups of 4; two row-strided b128 loads
        #pragma unroll
        for (int u = 0; u < 2; ++u) {
            int p = tid + u * 128;
            int kp = p >> 4, cg = p & 15;
            const float* w0 = W + (size_t)(k0 + 2 * kp) * N + n_blk + 4 * cg;
            float4 a0 = *(const float4*)w0;
            float4 a1 = *(const float4*)(w0 + N);
            Bpk[4 * cg + 0][kp] = pack_bf16(a0.x, a1.x);
            Bpk[4 * cg + 1][kp] = pack_bf16(a0.y, a1.y);
            Bpk[4 * cg + 2][kp] = pack_bf16(a0.z, a1.z);
            Bpk[4 * cg + 3][kp] = pack_bf16(a0.w, a1.w);
        }
        if (k0 + KT < K)  // speculative prefetch of next W slab
            __builtin_prefetch(W + (size_t)(k0 + KT) * N + n_blk + (tid & 63), 0, 1);
        __syncthreads();

        Frag fb;   // pair indices h*8 .. h*8+7 -> two ds_load_b128
        fb.q[0] = *(const u32x4*)&Bpk[ncol][h * 8];
        fb.q[1] = *(const u32x4*)&Bpk[ncol][h * 8 + 4];
        #pragma unroll
        for (int mt = 0; mt < 4; ++mt) {     // 4 WMMA per slab share fb
            Frag fa;
            fa.q[0] = *(const u32x4*)&Apk[mt * 16 + r16][h * 8];
            fa.q[1] = *(const u32x4*)&Apk[mt * 16 + r16][h * 8 + 4];
            acc[mt] = __builtin_amdgcn_wmma_f32_16x16x32_bf16(
                          false, fa.v, false, fb.v, (short)0, acc[mt], false, false);
        }
        __syncthreads();
    }

    int gn = n_blk + ncol;
    float bb = bias ? bias[gn] : 0.f;
    #pragma unroll
    for (int mt = 0; mt < 4; ++mt) {
        #pragma unroll
        for (int vv = 0; vv < 8; ++vv) {   // C: vgpr v -> row 8*h+v, col lane%16
            int gm = m_blk + mt * 16 + h * 8 + vv;
            float val = acc[mt][vv] + bb;
            if (ACT == 1) val = val > 0.f ? val : 0.f;
            else if (ACT == 2) val = tanhf(val);
            C[(size_t)gm * N + gn] = val;
        }
    }
}

template<int ACT, bool RIN>
static void gemm(const float* A, const float* W, const float* bias, float* C,
                 int M, int N, int K, hipStream_t s)
{
    dim3 g(M / BM, N / BN);
    hipLaunchKernelGGL(HIP_KERNEL_NAME(gemm_wmma_bf16<ACT, RIN>),
                       g, dim3(128), 0, s, A, W, bias, C, M, N, K);
}

// ---------------------------------------------------------------------------
// pack Wqk[64][64] = [ Wmq[64][32] | Wmk[64][32] ]  (so one N=64 GEMM does q&k)
// ---------------------------------------------------------------------------
__global__ __launch_bounds__(256)
void k_pack_qk(const float* __restrict__ Wmq, const float* __restrict__ Wmk,
               float* __restrict__ Wqk)
{
    int idx = blockIdx.x * 256 + threadIdx.x;       // 64*64 = 4096
    if (idx >= 4096) return;
    int r = idx >> 6, c = idx & 63;
    Wqk[idx] = (c < 32) ? Wmq[r * 32 + c] : Wmk[r * 32 + (c - 32)];
}

// ---------------------------------------------------------------------------
// GRU over T=48, hidden G=16, input 2; one thread per sequence (8192 seqs)
// ---------------------------------------------------------------------------
__global__ __launch_bounds__(256)
void k_gru(const float* __restrict__ glu, const float* __restrict__ timev,
           const float* __restrict__ Wx, const float* __restrict__ Wh,
           const float* __restrict__ bg, float* __restrict__ out)
{
    __shared__ float sWx[96], sWh[768], sb[48];
    for (int i = threadIdx.x; i < 96;  i += 256) sWx[i] = Wx[i];
    for (int i = threadIdx.x; i < 768; i += 256) sWh[i] = Wh[i];
    for (int i = threadIdx.x; i < 48;  i += 256) sb[i]  = bg[i];
    __syncthreads();

    int s = blockIdx.x * 256 + threadIdx.x;
    if (s >= 512 * 16) return;

    float hcur[16];
    #pragma unroll
    for (int i = 0; i < 16; ++i) hcur[i] = 0.f;
    const float* gp = glu   + (size_t)s * 48;
    const float* tp = timev + (size_t)s * 48;

    for (int t = 0; t < 48; ++t) {
        float x0 = gp[t], x1 = tp[t];
        float hn[16];
        #pragma unroll
        for (int i = 0; i < 16; ++i) {
            float gz = x0 * sWx[i]      + x1 * sWx[48 + i]      + sb[i];
            float gr = x0 * sWx[16 + i] + x1 * sWx[48 + 16 + i] + sb[16 + i];
            float gn = x0 * sWx[32 + i] + x1 * sWx[48 + 32 + i] + sb[32 + i];
            float ghz = 0.f, ghr = 0.f, ghn = 0.f;
            #pragma unroll
            for (int k2 = 0; k2 < 16; ++k2) {
                float hk = hcur[k2];
                ghz += hk * sWh[k2 * 48 + i];
                ghr += hk * sWh[k2 * 48 + 16 + i];
                ghn += hk * sWh[k2 * 48 + 32 + i];
            }
            float z = 1.f / (1.f + __expf(-(gz + ghz)));
            float r = 1.f / (1.f + __expf(-(gr + ghr)));
            float n = tanhf(gn + r * ghn);
            hn[i] = (1.f - z) * n + z * hcur[i];
        }
        #pragma unroll
        for (int i = 0; i < 16; ++i) hcur[i] = hn[i];
    }
    for (int i = 0; i < 16; ++i) out[(size_t)s * 16 + i] = hcur[i];
}

// ---------------------------------------------------------------------------
// glu attn-encoder + fused static layer-2 + broadcast -> patient_rep [B,16,64]
// ---------------------------------------------------------------------------
__global__ __launch_bounds__(128)
void k_glu_attn(const float* __restrict__ glu_enc, const float* __restrict__ Wg1,
                const float* __restrict__ bg1, const float* __restrict__ Wgq,
                const float* __restrict__ Wgk,
                const float* __restrict__ h1, const float* __restrict__ Ws2,
                const float* __restrict__ bs2,
                float* __restrict__ patient)
{
    const int b = blockIdx.x, tid = threadIdx.x;
    __shared__ float xe[16][16], hh[16][32], qq[16][16], kk[16][16], aa[16][16];
    __shared__ float sh1[64], s2row[32];

    for (int i = tid; i < 256; i += 128) xe[i >> 4][i & 15] = glu_enc[b * 256 + i];
    if (tid < 64) sh1[tid] = h1[b * 64 + tid];
    __syncthreads();
    if (tid < 32) {   // fused static layer 2: s2row = relu(h1[b] @ Ws2 + bs2)
        float acc = bs2[tid];
        for (int j = 0; j < 64; ++j) acc += sh1[j] * Ws2[j * 32 + tid];
        s2row[tid] = fmaxf(acc, 0.f);
    }
    for (int i = tid; i < 512; i += 128) {
        int r = i >> 5, c = i & 31;
        float acc = bg1[c];
        for (int j = 0; j < 16; ++j) acc += xe[r][j] * Wg1[j * 32 + c];
        hh[r][c] = tanhf(acc);
    }
    __syncthreads();
    for (int i = tid; i < 512; i += 128) {
        int r = (i >> 4) & 15, c = i & 15;
        const float* Wp = (i < 256) ? Wgq : Wgk;
        float acc = 0.f;
        for (int j = 0; j < 32; ++j) acc += hh[r][j] * Wp[j * 16 + c];
        if (i < 256) qq[r][c] = acc; else kk[r][c] = acc;
    }
    __syncthreads();
    for (int i = tid; i < 256; i += 128) {
        int r = i >> 4, c = i & 15;
        float acc = 0.f;
        for (int j = 0; j < 16; ++j) acc += qq[r][j] * kk[c][j];
        aa[r][c] = acc * 0.25f;  // 1/sqrt(16)
    }
    __syncthreads();
    if (tid < 16) {
        float mx = -1e30f;
        for (int j = 0; j < 16; ++j) mx = fmaxf(mx, aa[tid][j]);
        float sm = 0.f;
        for (int j = 0; j < 16; ++j) { float e = __expf(aa[tid][j] - mx); aa[tid][j] = e; sm += e; }
        float inv = 1.f / sm;
        for (int j = 0; j < 16; ++j) aa[tid][j] *= inv;
    }
    __syncthreads();
    for (int i = tid; i < 512; i += 128) {
        int r = i >> 5, c = i & 31;
        float acc = 0.f;
        for (int j = 0; j < 16; ++j) acc += aa[r][j] * hh[j][c];
        patient[(size_t)b * 1024 + r * 64 + c]      = acc;       // glu_rep
        patient[(size_t)b * 1024 + r * 64 + 32 + c] = s2row[c];  // static
    }
}

// ---------------------------------------------------------------------------
// med attn-encoder: qk [B*16,64] (q=cols 0..31, k=cols 32..63), h [B*16,64]
// ---------------------------------------------------------------------------
__global__ __launch_bounds__(128)
void k_attn_med(const float* __restrict__ hmed, const float* __restrict__ qkmed,
                float* __restrict__ med_rep)
{
    const int b = blockIdx.x, tid = threadIdx.x;
    __shared__ float aa[15][15];
    const float scale = 0.17677669529663687f; // 1/sqrt(32)
    for (int i = tid; i < 225; i += 128) {
        int r = i / 15, c = i % 15;
        const float* qp = qkmed + (size_t)(b * 16 + r) * 64;       // q part
        const float* kp = qkmed + (size_t)(b * 16 + c) * 64 + 32;  // k part
        float acc = 0.f;
        for (int j = 0; j < 32; ++j) acc += qp[j] * kp[j];
        aa[r][c] = acc * scale;
    }
    __syncthreads();
    if (tid < 15) {
        float mx = -1e30f;
        for (int j = 0; j < 15; ++j) mx = fmaxf(mx, aa[tid][j]);
        float sm = 0.f;
        for (int j = 0; j < 15; ++j) { float e = __expf(aa[tid][j] - mx); aa[tid][j] = e; sm += e; }
        float inv = 1.f / sm;
        for (int j = 0; j < 15; ++j) aa[tid][j] *= inv;
    }
    __syncthreads();
    for (int i = tid; i < 16 * 64; i += 128) {
        int r = i >> 6, c = i & 63;
        float acc = 0.f;
        if (r < 15)
            for (int j = 0; j < 15; ++j)
                acc += aa[r][j] * hmed[(size_t)(b * 16 + j) * 64 + c];
        med_rep[(size_t)(b * 16 + r) * 64 + c] = acc;  // row 15 zeroed
    }
}

// ---------------------------------------------------------------------------
// MHA-1 attention core: q len 15, kv len 16, H=4, dh=16 -> o1 [B*16(pad),64]
// ---------------------------------------------------------------------------
__global__ __launch_bounds__(64)
void k_mha1(const float* __restrict__ qh, const float* __restrict__ kh,
            const float* __restrict__ vh, float* __restrict__ o1)
{
    const int b = blockIdx.x, hd = blockIdx.y, tid = threadIdx.x;
    __shared__ float kk[16][16], vv[16][16], aa[15][16];
    for (int i = tid; i < 256; i += 64) {
        int r = i >> 4, d = i & 15;
        kk[r][d] = kh[(size_t)(b * 16 + r) * 64 + hd * 16 + d];
        vv[r][d] = vh[(size_t)(b * 16 + r) * 64 + hd * 16 + d];
    }
    __syncthreads();
    for (int i = tid; i < 240; i += 64) {
        int r = i >> 4, c = i & 15;
        const float* qp = qh + (size_t)(b * 16 + r) * 64 + hd * 16;
        float acc = 0.f;
        for (int j = 0; j < 16; ++j) acc += qp[j] * kk[c][j];
        aa[r][c] = acc * 0.25f;  // 1/sqrt(16)
    }
    __syncthreads();
    if (tid < 15) {
        float mx = -1e30f;
        for (int j = 0; j < 16; ++j) mx = fmaxf(mx, aa[tid][j]);
        float sm = 0.f;
        for (int j = 0; j < 16; ++j) { float e = __expf(aa[tid][j] - mx); aa[tid][j] = e; sm += e; }
        float inv = 1.f / sm;
        for (int j = 0; j < 16; ++j) aa[tid][j] *= inv;
    }
    __syncthreads();
    for (int i = tid; i < 256; i += 64) {
        int r = i >> 4, d = i & 15;
        float acc = 0.f;
        if (r < 15)
            for (int j = 0; j < 16; ++j) acc += aa[r][j] * vv[j][d];
        o1[(size_t)(b * 16 + r) * 64 + hd * 16 + d] = acc;  // row 15 zeroed
    }
}

// ---------------------------------------------------------------------------
// drug memory: (ehr@Wl1+bl1)+(time_adj@Wl2+bl2)-(ddi@Wl3+bl3) -> [192,64]
// ---------------------------------------------------------------------------
__global__ __launch_bounds__(256)
void k_drugmem(const float* __restrict__ ehr, const float* __restrict__ ddi,
               const float* __restrict__ tadj,
               const float* __restrict__ Wl1, const float* __restrict__ bl1,
               const float* __restrict__ Wl2, const float* __restrict__ bl2,
               const float* __restrict__ Wl3, const float* __restrict__ bl3,
               float* __restrict__ out)
{
    int idx = blockIdx.x * 256 + threadIdx.x;
    if (idx >= 192 * 64) return;
    int m = idx >> 6, e = idx & 63;
    float a1 = 0.f, a2 = 0.f, a3 = 0.f;
    for (int k2 = 0; k2 < 192; ++k2) {
        a1 += ehr[m * 192 + k2]  * Wl1[k2 * 64 + e];
        a2 += tadj[m * 192 + k2] * Wl2[k2 * 64 + e];
        a3 += ddi[m * 192 + k2]  * Wl3[k2 * 64 + e];
    }
    out[idx] = (a1 + bl1[e]) + (a2 + bl2[e]) - (a3 + bl3[e]);
}

// ---------------------------------------------------------------------------
// MHA-2 attention core: q len 192 (shared qh2), kv len 15 -> o2 [B,192,64]
// ---------------------------------------------------------------------------
__global__ __launch_bounds__(192)
void k_mha2(const float* __restrict__ qh2, const float* __restrict__ kh2,
            const float* __restrict__ vh2, float* __restrict__ o2)
{
    const int b = blockIdx.x, hd = blockIdx.y, m = threadIdx.x;
    __shared__ float kk[15][16], vv[15][16];
    for (int i = m; i < 240; i += 192) {
        int r = i >> 4, d = i & 15;
        kk[r][d] = kh2[(size_t)(b * 16 + r) * 64 + hd * 16 + d];
        vv[r][d] = vh2[(size_t)(b * 16 + r) * 64 + hd * 16 + d];
    }
    __syncthreads();
    float q[16];
    #pragma unroll
    for (int d = 0; d < 16; ++d) q[d] = qh2[(size_t)m * 64 + hd * 16 + d];
    float sc[15];
    float mx = -1e30f;
    #pragma unroll
    for (int j = 0; j < 15; ++j) {
        float acc = 0.f;
        #pragma unroll
        for (int d = 0; d < 16; ++d) acc += q[d] * kk[j][d];
        sc[j] = acc * 0.25f;  // 1/sqrt(16)
        mx = fmaxf(mx, sc[j]);
    }
    float sm = 0.f;
    #pragma unroll
    for (int j = 0; j < 15; ++j) { sc[j] = __expf(sc[j] - mx); sm += sc[j]; }
    float inv = 1.f / sm;
    float* op = o2 + (size_t)b * 12288 + (size_t)m * 64 + hd * 16;
    #pragma unroll
    for (int d = 0; d < 16; ++d) {
        float acc = 0.f;
        #pragma unroll
        for (int j = 0; j < 15; ++j) acc += sc[j] * vv[j][d];
        op[d] = acc * inv;
    }
}

// ---------------------------------------------------------------------------
extern "C" void kernel_launch(void* const* d_in, const int* in_sizes, int n_in,
                              void* d_out, int out_size, void* d_ws, size_t ws_size,
                              hipStream_t stream)
{
    (void)in_sizes; (void)n_in; (void)out_size; (void)ws_size;
    const float* lab   = (const float*)d_in[0];
    const float* glu   = (const float*)d_in[1];
    const float* timev = (const float*)d_in[2];
    const float* med   = (const float*)d_in[3];
    const float* ehr   = (const float*)d_in[4];
    const float* ddi   = (const float*)d_in[5];
    const float* tadj  = (const float*)d_in[6];
    const float* Ws1 = (const float*)d_in[7];  const float* bs1 = (const float*)d_in[8];
    const float* Ws2 = (const float*)d_in[9];  const float* bs2 = (const float*)d_in[10];
    const float* Wl1 = (const float*)d_in[11]; const float* bl1 = (const float*)d_in[12];
    const float* Wl2 = (const float*)d_in[13]; const float* bl2 = (const float*)d_in[14];
    const float* Wl3 = (const float*)d_in[15]; const float* bl3 = (const float*)d_in[16];
    const float* Wm1 = (const float*)d_in[17]; const float* bm1 = (const float*)d_in[18];
    const float* Wmq = (const float*)d_in[19]; const float* Wmk = (const float*)d_in[20];
    const float* Wg1 = (const float*)d_in[21]; const float* bg1 = (const float*)d_in[22];
    const float* Wgq = (const float*)d_in[23]; const float* Wgk = (const float*)d_in[24];
    const float* gWx = (const float*)d_in[25]; const float* gWh = (const float*)d_in[26];
    const float* gb  = (const float*)d_in[27];
    const float* m1Wq = (const float*)d_in[28]; const float* m1Wk = (const float*)d_in[29];
    const float* m1Wv = (const float*)d_in[30]; const float* m1Wo = (const float*)d_in[31];
    const float* m2Wq = (const float*)d_in[32]; const float* m2Wk = (const float*)d_in[33];
    const float* m2Wv = (const float*)d_in[34]; const float* m2Wo = (const float*)d_in[35];
    const float* Wo1 = (const float*)d_in[36]; const float* bo1 = (const float*)d_in[37];
    const float* Wo2 = (const float*)d_in[38]; const float* bo2 = (const float*)d_in[39];
    float* out = (float*)d_out;

    // workspace layout (fp32 elements)
    float* ws = (float*)d_ws;
    size_t off = 0;
    auto alloc = [&](size_t n) { float* p = ws + off; off += n; return p; };
    float* h1      = alloc(512 * 64);
    float* glu_enc = alloc(8192 * 16);
    float* patient = alloc(8192 * 64);
    float* h_med   = alloc(8192 * 64);
    float* Wqk     = alloc(64 * 64);
    float* qk_med  = alloc(8192 * 64);
    float* med_rep = alloc(8192 * 64);
    float* qh1     = alloc(8192 * 64);
    float* kh1     = alloc(8192 * 64);
    float* vh1     = alloc(8192 * 64);
    float* o1      = alloc(8192 * 64);
    float* E_en    = alloc(8192 * 64);
    float* drug    = alloc(192 * 64);
    float* qh2     = alloc(192 * 64);
    float* kh2     = alloc(8192 * 64);
    float* vh2     = alloc(8192 * 64);
    float* o2      = alloc((size_t)98304 * 64);
    float* E_de    = alloc((size_t)98304 * 64);
    float* fc1     = alloc(512 * 1536);

    // static encoder layer 1 (layer 2 fused into k_glu_attn)
    gemm<1, false>(lab, Ws1, bs1, h1, 512, 64, 256, stream);

    // GRU glucose encoder
    hipLaunchKernelGGL(k_gru, dim3(32), dim3(256), 0, stream,
                       glu, timev, gWx, gWh, gb, glu_enc);

    // glucose attention encoder + fused static layer 2 -> patient_rep
    hipLaunchKernelGGL(k_glu_attn, dim3(512), dim3(128), 0, stream,
                       glu_enc, Wg1, bg1, Wgq, Wgk, h1, Ws2, bs2, patient);

    // med attention encoder (row v=15 is pad, never consumed)
    gemm<2, false>(med, Wm1, bm1, h_med, 8192, 64, 192, stream);
    hipLaunchKernelGGL(k_pack_qk, dim3(16), dim3(256), 0, stream, Wmq, Wmk, Wqk);
    gemm<0, false>(h_med, Wqk, nullptr, qk_med, 8192, 64, 64, stream);  // q|k fused
    hipLaunchKernelGGL(k_attn_med, dim3(512), dim3(128), 0, stream,
                       h_med, qk_med, med_rep);

    // MHA 1: queries from med_rep (len 15), keys/values from patient_rep (len 16)
    gemm<0, false>(med_rep, m1Wq, nullptr, qh1, 8192, 64, 64, stream);
    gemm<0, false>(patient, m1Wk, nullptr, kh1, 8192, 64, 64, stream);
    gemm<0, false>(patient, m1Wv, nullptr, vh1, 8192, 64, 64, stream);
    hipLaunchKernelGGL(k_mha1, dim3(512, 4), dim3(64), 0, stream, qh1, kh1, vh1, o1);
    gemm<0, false>(o1, m1Wo, nullptr, E_en, 8192, 64, 64, stream);

    // drug memory (shared across batch) + its shared query projection
    hipLaunchKernelGGL(k_drugmem, dim3(48), dim3(256), 0, stream,
                       ehr, ddi, tadj, Wl1, bl1, Wl2, bl2, Wl3, bl3, drug);
    gemm<0, false>(drug, m2Wq, nullptr, qh2, 192, 64, 64, stream);

    // MHA 2: queries = drug memory (192), keys/values from E_en (len 15)
    gemm<0, false>(E_en, m2Wk, nullptr, kh2, 8192, 64, 64, stream);
    gemm<0, false>(E_en, m2Wv, nullptr, vh2, 8192, 64, 64, stream);
    hipLaunchKernelGGL(k_mha2, dim3(512, 4), dim3(192), 0, stream, qh2, kh2, vh2, o2);
    gemm<0, false>(o2, m2Wo, nullptr, E_de, 98304, 64, 64, stream);

    // head: relu(E_de) @ Wo1 + bo1 -> relu -> @ Wo2 + bo2
    gemm<1, true >(E_de, Wo1, bo1, fc1, 512, 1536, 12288, stream);  // dominant GEMM
    gemm<0, false>(fc1, Wo2, bo2, out, 512, 192, 1536, stream);
}
